// Local_Transfor_46712064311409
// MI455X (gfx1250) — matrix operand order repaired and checked by reading, not compiled, not run on hardware
//
#include <hip/hip_runtime.h>

// ---------------------------------------------------------------------------
// CDNA5 (gfx1250, wave32) fused 3-block MHA pipeline with bf16 WMMA.
// ---------------------------------------------------------------------------

typedef __attribute__((ext_vector_type(16))) __bf16 v16bf;
typedef __attribute__((ext_vector_type(8)))  __bf16 v8bf;
typedef __attribute__((ext_vector_type(4)))  __bf16 v4bf;
typedef __attribute__((ext_vector_type(8)))  float  v8f;

#define D_MODEL 768
#define SEQ_L   1024
#define BATCH   8
#define HEADS   12
#define DK      64
#define NROWS   (BATCH * SEQ_L)                 // 8192
#define NEL     ((size_t)NROWS * D_MODEL)       // 6,291,456 elements

union VA {
    v16bf v;
    v8bf  h8[2];
    v4bf  h4[4];
};

__device__ inline v8f wmma_bf16(v16bf a, v16bf b, v8f c) {
    // D = A(16x32 bf16) * B(32x16 bf16) + C(16x16 f32)
    return __builtin_amdgcn_wmma_f32_16x16x32_bf16(
        /*neg_a=*/false, a, /*neg_b=*/false, b,
        /*c_mod=*/(short)0, c, /*reuse_a=*/false, /*reuse_b=*/false);
}

// ---------------------------------------------------------------------------
// f32 -> bf16 row-major convert
// ---------------------------------------------------------------------------
__global__ __launch_bounds__(256)
void cvt_f32_bf16(const float* __restrict__ in, __bf16* __restrict__ out, int n) {
    int i = (blockIdx.x * blockDim.x + threadIdx.x) * 4;
    if (i + 3 < n) {
        float4 f = *(const float4*)(in + i);
        out[i + 0] = (__bf16)f.x;
        out[i + 1] = (__bf16)f.y;
        out[i + 2] = (__bf16)f.z;
        out[i + 3] = (__bf16)f.w;
    }
}

// ---------------------------------------------------------------------------
// Weight convert + transpose: Wt[n*D + k] = bf16(W[k*D + n])  (D x D)
// Makes every WMMA B-fragment load contiguous.
// ---------------------------------------------------------------------------
__global__ __launch_bounds__(256)
void wt_transpose_bf16(const float* __restrict__ w, __bf16* __restrict__ wt) {
    int i = blockIdx.x * blockDim.x + threadIdx.x;   // coalesced read
    int k = i / D_MODEL;
    int n = i - k * D_MODEL;
    wt[(size_t)n * D_MODEL + k] = (__bf16)w[i];
}

// ---------------------------------------------------------------------------
// GEMM: C[N,D] = A_bf16[N,D] @ W (via Wt bf16) + bias
//   mode 0: write bf16 row-major (Q / K projections)
//   mode 1: write bf16 head-transposed Vt[b][h][d][l]  (V projection)
//   mode 2: write f32 row-major to outf, optional bf16 row-major copy to outbf
// Block = 256 threads (8 waves). Block tile 64x64, wave tile 16x32.
// A and Wt tiles staged through LDS (padded pitch = 40 halves).
// ---------------------------------------------------------------------------
__global__ __launch_bounds__(256)
void gemm_bf16_wmma(const __bf16* __restrict__ A, const __bf16* __restrict__ Wt,
                    const float* __restrict__ bias, int mode,
                    float* __restrict__ outf, __bf16* __restrict__ outbf) {
    __shared__ __bf16 sA[64 * 40];
    __shared__ __bf16 sB[64 * 40];

    const int tid    = threadIdx.x;
    const int lane   = tid & 31;
    const int wid    = tid >> 5;
    const int laneHi = lane >> 4;
    const int lane15 = lane & 15;
    const int waveRow = wid >> 1;       // 0..3
    const int waveCol = wid & 1;        // 0..1
    const int rowOff = blockIdx.x * 64;
    const int colOff = blockIdx.y * 64;
    const int lr = tid >> 2;            // 0..63 staged row
    const int lc = (tid & 3) * 8;       // staged col group (halves)

    v8f acc0 = {0.f, 0.f, 0.f, 0.f, 0.f, 0.f, 0.f, 0.f};
    v8f acc1 = {0.f, 0.f, 0.f, 0.f, 0.f, 0.f, 0.f, 0.f};

    for (int kb = 0; kb < D_MODEL; kb += 32) {
        // ---- stage A and Wt tiles into LDS (b128 global loads, b64 DS stores)
        {
            VA t;
            t.h8[0] = *(const v8bf*)(A + (size_t)(rowOff + lr) * D_MODEL + kb + lc);
            *(v4bf*)&sA[lr * 40 + lc]     = t.h4[0];
            *(v4bf*)&sA[lr * 40 + lc + 4] = t.h4[1];
            t.h8[0] = *(const v8bf*)(Wt + (size_t)(colOff + lr) * D_MODEL + kb + lc);
            *(v4bf*)&sB[lr * 40 + lc]     = t.h4[0];
            *(v4bf*)&sB[lr * 40 + lc + 4] = t.h4[1];
            if (kb + 32 < D_MODEL)
                __builtin_prefetch(A + (size_t)(rowOff + lr) * D_MODEL + kb + 32 + lc, 0, 1);
        }
        __syncthreads();

        // ---- A fragment (16x32): lanes 0-15 rows M=0..15; K runs split by lane half
        VA af;
        const int arow = waveRow * 16 + lane15;
        const int k0   = laneHi * 16;
        af.h4[0] = *(const v4bf*)&sA[arow * 40 + k0];
        af.h4[1] = *(const v4bf*)&sA[arow * 40 + k0 + 4];
        af.h4[2] = *(const v4bf*)&sA[arow * 40 + k0 + 8];
        af.h4[3] = *(const v4bf*)&sA[arow * 40 + k0 + 12];

#pragma unroll
        for (int sub = 0; sub < 2; ++sub) {
            // B fragment (32x16): lane = column n, K = laneHi*16 + j
            VA bfr;
            const int n  = waveCol * 32 + sub * 16 + lane15;
            const int kk = laneHi * 16;
            bfr.h4[0] = *(const v4bf*)&sB[n * 40 + kk];
            bfr.h4[1] = *(const v4bf*)&sB[n * 40 + kk + 4];
            bfr.h4[2] = *(const v4bf*)&sB[n * 40 + kk + 8];
            bfr.h4[3] = *(const v4bf*)&sB[n * 40 + kk + 12];
            if (sub == 0) acc0 = wmma_bf16(af.v, bfr.v, acc0);
            else          acc1 = wmma_bf16(af.v, bfr.v, acc1);
        }
        __syncthreads();
    }

    // ---- epilogue: C layout row = r + 8*laneHi, col = lane15
#pragma unroll
    for (int sub = 0; sub < 2; ++sub) {
        v8f a = sub ? acc1 : acc0;
        const int col  = colOff + waveCol * 32 + sub * 16 + lane15;
        const float bv = bias[col];
#pragma unroll
        for (int r = 0; r < 8; ++r) {
            const int row = rowOff + waveRow * 16 + laneHi * 8 + r;
            const float v = a[r] + bv;
            if (mode == 0) {
                outbf[(size_t)row * D_MODEL + col] = (__bf16)v;
            } else if (mode == 1) {
                const int bi = row >> 10;      // row / SEQ_L
                const int l  = row & 1023;
                const int h  = col >> 6;       // col / DK
                const int d  = col & 63;
                outbf[(((size_t)bi * HEADS + h) * DK + d) * SEQ_L + l] = (__bf16)v;
            } else {
                outf[(size_t)row * D_MODEL + col] = v;
                if (outbf) outbf[(size_t)row * D_MODEL + col] = (__bf16)v;
            }
        }
    }
}

// ---------------------------------------------------------------------------
// Flash attention per (b, h, 16-row q tile), one wave per tile, block = 4 waves.
// S = (Q Kt)/8 via 2x2 WMMA per 32-key tile, online softmax, P@V via 4 WMMA
// with head-transposed V.  Mask (optional): 0 -> -1e9 before softmax.
// ---------------------------------------------------------------------------
__global__ __launch_bounds__(128)
void flash_attn_wmma(const __bf16* __restrict__ Q, const __bf16* __restrict__ K,
                     const __bf16* __restrict__ Vt, const int* __restrict__ mask,
                     __bf16* __restrict__ O) {
    __shared__ __bf16 sP[4][16 * 40];   // per-wave P tile (16 x 32, padded)

    const int lane   = threadIdx.x & 31;
    const int wid    = threadIdx.x >> 5;
    const int laneHi = lane >> 4;
    const int lane15 = lane & 15;

    const int task = blockIdx.x * 4 + wid;        // 6144 total
    const int qt   = task & 63;
    const int rest = task >> 6;
    const int h    = rest % HEADS;
    const int b    = rest / HEADS;
    const int q0   = qt * 16;

    // ---- Q fragments (dk=64 -> 2 x 16x32 A fragments), contiguous 16B loads
    VA qf[2];
    {
        const __bf16* qp = Q + (size_t)(b * SEQ_L + q0 + lane15) * D_MODEL + h * DK;
#pragma unroll
        for (int kb2 = 0; kb2 < 2; ++kb2) {
            qf[kb2].h8[0] = *(const v8bf*)(qp + kb2 * 32 + laneHi * 16);
            qf[kb2].h8[1] = *(const v8bf*)(qp + kb2 * 32 + laneHi * 16 + 8);
        }
    }

    v8f o[4];
    float m[8], lsum[8];
#pragma unroll
    for (int sl = 0; sl < 4; ++sl) o[sl] = (v8f){0.f, 0.f, 0.f, 0.f, 0.f, 0.f, 0.f, 0.f};
#pragma unroll
    for (int r = 0; r < 8; ++r) { m[r] = -3.0e38f; lsum[r] = 0.f; }

    for (int kt = 0; kt < SEQ_L / 32; ++kt) {
        const int kbase = kt * 32;

        // ---- S = Q @ K^T * (1/sqrt(dk)), 32 keys = two 16-col halves
        v8f s[2];
#pragma unroll
        for (int half = 0; half < 2; ++half) {
            v8f c = {0.f, 0.f, 0.f, 0.f, 0.f, 0.f, 0.f, 0.f};
            const __bf16* kp =
                K + (size_t)(b * SEQ_L + kbase + half * 16 + lane15) * D_MODEL + h * DK;
#pragma unroll
            for (int kb2 = 0; kb2 < 2; ++kb2) {
                VA bfr;   // B frag: lane = key col, K = dk (contiguous in memory)
                bfr.h8[0] = *(const v8bf*)(kp + kb2 * 32 + laneHi * 16);
                bfr.h8[1] = *(const v8bf*)(kp + kb2 * 32 + laneHi * 16 + 8);
                c = wmma_bf16(qf[kb2].v, bfr.v, c);
            }
            s[half] = c * 0.125f;
        }

        // ---- optional additive mask (0 => -1e9)
        if (mask) {
#pragma unroll
            for (int half = 0; half < 2; ++half) {
#pragma unroll
                for (int r = 0; r < 8; ++r) {
                    const int qr  = q0 + laneHi * 8 + r;
                    const int key = kbase + half * 16 + lane15;
                    if (mask[((size_t)b * SEQ_L + qr) * SEQ_L + key] == 0)
                        s[half][r] = -1e9f;
                }
            }
        }

        // ---- online softmax (row stats replicated across each 16-lane group)
#pragma unroll
        for (int r = 0; r < 8; ++r) {
            float t = fmaxf(s[0][r], s[1][r]);
#pragma unroll
            for (int off = 8; off > 0; off >>= 1)
                t = fmaxf(t, __shfl_xor(t, off, 32));
            const float mnew = fmaxf(m[r], t);
            const float corr = __expf(m[r] - mnew);
            const float p0 = __expf(s[0][r] - mnew);
            const float p1 = __expf(s[1][r] - mnew);
            s[0][r] = p0; s[1][r] = p1;
            float rs = p0 + p1;
#pragma unroll
            for (int off = 8; off > 0; off >>= 1)
                rs += __shfl_xor(rs, off, 32);
            lsum[r] = lsum[r] * corr + rs;
            m[r] = mnew;
            o[0][r] *= corr; o[1][r] *= corr; o[2][r] *= corr; o[3][r] *= corr;
        }

        // ---- P (C layout) -> LDS -> reread as A fragment (intra-wave, DScnt-ordered)
#pragma unroll
        for (int half = 0; half < 2; ++half)
#pragma unroll
            for (int r = 0; r < 8; ++r)
                sP[wid][(laneHi * 8 + r) * 40 + half * 16 + lane15] = (__bf16)s[half][r];

        VA pf;
        pf.h4[0] = *(const v4bf*)&sP[wid][lane15 * 40 + laneHi * 16];
        pf.h4[1] = *(const v4bf*)&sP[wid][lane15 * 40 + laneHi * 16 + 4];
        pf.h4[2] = *(const v4bf*)&sP[wid][lane15 * 40 + laneHi * 16 + 8];
        pf.h4[3] = *(const v4bf*)&sP[wid][lane15 * 40 + laneHi * 16 + 12];

        // ---- O += P @ V  (V head-transposed => contiguous B-fragment loads)
#pragma unroll
        for (int sl = 0; sl < 4; ++sl) {
            const __bf16* vp =
                Vt + ((size_t)(b * HEADS + h) * DK + sl * 16 + lane15) * SEQ_L +
                kbase + laneHi * 16;
            VA bfr;
            bfr.h8[0] = *(const v8bf*)vp;
            bfr.h8[1] = *(const v8bf*)(vp + 8);
            o[sl] = wmma_bf16(pf.v, bfr.v, o[sl]);
        }
    }

    // ---- normalize and write bf16 row-major attention output
#pragma unroll
    for (int sl = 0; sl < 4; ++sl) {
        const int col = h * DK + sl * 16 + lane15;
#pragma unroll
        for (int r = 0; r < 8; ++r) {
            const int row = b * SEQ_L + q0 + laneHi * 8 + r;
            O[(size_t)row * D_MODEL + col] = (__bf16)(o[sl][r] / lsum[r]);
        }
    }
}

// ---------------------------------------------------------------------------
// Dual-LayerNorm fuse: out = LN(z)*g1+b1 + LN(x)*g2+b2, bf16 output.
// One block per row of 768.
// ---------------------------------------------------------------------------
__global__ __launch_bounds__(256)
void ln_fuse_kernel(const float* __restrict__ z, const float* __restrict__ x,
                    const float* __restrict__ g1, const float* __restrict__ b1,
                    const float* __restrict__ g2, const float* __restrict__ b2,
                    __bf16* __restrict__ out) {
    __shared__ float red[8][4];
    const int row = blockIdx.x;
    const float* zr = z + (size_t)row * D_MODEL;
    const float* xr = x + (size_t)row * D_MODEL;

    float sz = 0.f, szz = 0.f, sx = 0.f, sxx = 0.f;
    for (int c = threadIdx.x; c < D_MODEL; c += 256) {
        const float a = zr[c], v = xr[c];
        sz += a; szz += a * a; sx += v; sxx += v * v;
    }
    const int lane = threadIdx.x & 31, wid = threadIdx.x >> 5;
#pragma unroll
    for (int off = 16; off > 0; off >>= 1) {
        sz  += __shfl_xor(sz,  off, 32);
        szz += __shfl_xor(szz, off, 32);
        sx  += __shfl_xor(sx,  off, 32);
        sxx += __shfl_xor(sxx, off, 32);
    }
    if (lane == 0) { red[wid][0] = sz; red[wid][1] = szz; red[wid][2] = sx; red[wid][3] = sxx; }
    __syncthreads();
    sz = szz = sx = sxx = 0.f;
#pragma unroll
    for (int i = 0; i < 8; ++i) {
        sz += red[i][0]; szz += red[i][1]; sx += red[i][2]; sxx += red[i][3];
    }
    const float inv = 1.0f / (float)D_MODEL;
    const float m1 = sz * inv, v1 = szz * inv - m1 * m1;
    const float m2 = sx * inv, v2 = sxx * inv - m2 * m2;
    const float r1 = rsqrtf(v1 + 1e-5f), r2 = rsqrtf(v2 + 1e-5f);
    for (int c = threadIdx.x; c < D_MODEL; c += 256) {
        const float v = (zr[c] - m1) * r1 * g1[c] + b1[c] +
                        (xr[c] - m2) * r2 * g2[c] + b2[c];
        out[(size_t)row * D_MODEL + c] = (__bf16)v;
    }
}

// ---------------------------------------------------------------------------
// Host orchestration
// ---------------------------------------------------------------------------
extern "C" void kernel_launch(void* const* d_in, const int* in_sizes, int n_in,
                              void* d_out, int out_size, void* d_ws, size_t ws_size,
                              hipStream_t stream) {
    (void)in_sizes; (void)n_in; (void)out_size; (void)ws_size;

    const float* w_feat  = (const float*)d_in[0];
    const float* s_feat  = (const float*)d_in[1];
    const float* v_feat1 = (const float*)d_in[2];
    const float* v_feat2 = (const float*)d_in[3];
    const int*   amask   = (const int*)d_in[4];
    const float* Wq = (const float*)d_in[5];
    const float* Wk = (const float*)d_in[6];
    const float* Wv = (const float*)d_in[7];
    const float* Wo = (const float*)d_in[8];
    const float* bq = (const float*)d_in[9];
    const float* bk = (const float*)d_in[10];
    const float* bv = (const float*)d_in[11];
    const float* bo = (const float*)d_in[12];
    const float* ln1g = (const float*)d_in[13];
    const float* ln1b = (const float*)d_in[14];
    const float* ln2g = (const float*)d_in[15];
    const float* ln2b = (const float*)d_in[16];
    float* out = (float*)d_out;

    const size_t NE = NEL;
    const int    DD = D_MODEL * D_MODEL;

    char* ws = (char*)d_ws;
    __bf16* Qbf     = (__bf16*)ws; ws += NE * 2;
    __bf16* Kbf     = (__bf16*)ws; ws += NE * 2;
    __bf16* Vtbf    = (__bf16*)ws; ws += NE * 2;
    __bf16* Obf     = (__bf16*)ws; ws += NE * 2;
    __bf16* bfA     = (__bf16*)ws; ws += NE * 2;
    __bf16* bfKV    = (__bf16*)ws; ws += NE * 2;
    __bf16* zwsbf   = (__bf16*)ws; ws += NE * 2;
    __bf16* fusedbf = (__bf16*)ws; ws += NE * 2;
    float*  zvw     = (float*)ws;  ws += NE * 4;
    __bf16* Wt      = (__bf16*)ws; ws += (size_t)DD * 2;

    const dim3 bGemm(256), gGemm(NROWS / 64, D_MODEL / 64);
    const dim3 bAttn(128), gAttn(BATCH * HEADS * (SEQ_L / 16) / 4);
    const int  gCvt = (int)(NE / 1024);
    const int  gWt  = DD / 256;

    // ---- inputs -> bf16
    cvt_f32_bf16<<<gCvt, 256, 0, stream>>>(w_feat, bfA, (int)NE);
    cvt_f32_bf16<<<gCvt, 256, 0, stream>>>(s_feat, bfKV, (int)NE);

    // ================= MHA block 0: z_ws = MHA(w_feat, s_feat, s_feat) ======
    wt_transpose_bf16<<<gWt, 256, 0, stream>>>(Wq + 0 * DD, Wt);
    gemm_bf16_wmma<<<gGemm, bGemm, 0, stream>>>(bfA,  Wt, bq + 0 * D_MODEL, 0, nullptr, Qbf);
    wt_transpose_bf16<<<gWt, 256, 0, stream>>>(Wk + 0 * DD, Wt);
    gemm_bf16_wmma<<<gGemm, bGemm, 0, stream>>>(bfKV, Wt, bk + 0 * D_MODEL, 0, nullptr, Kbf);
    wt_transpose_bf16<<<gWt, 256, 0, stream>>>(Wv + 0 * DD, Wt);
    gemm_bf16_wmma<<<gGemm, bGemm, 0, stream>>>(bfKV, Wt, bv + 0 * D_MODEL, 1, nullptr, Vtbf);
    flash_attn_wmma<<<gAttn, bAttn, 0, stream>>>(Qbf, Kbf, Vtbf, nullptr, Obf);
    wt_transpose_bf16<<<gWt, 256, 0, stream>>>(Wo + 0 * DD, Wt);
    gemm_bf16_wmma<<<gGemm, bGemm, 0, stream>>>(Obf, Wt, bo + 0 * D_MODEL, 2, out + NE, zwsbf);

    // ================= MHA block 1: z_vw = MHA(v_feat2, z_ws, z_ws, mask) ===
    cvt_f32_bf16<<<gCvt, 256, 0, stream>>>(v_feat2, bfA, (int)NE);
    wt_transpose_bf16<<<gWt, 256, 0, stream>>>(Wq + 1 * DD, Wt);
    gemm_bf16_wmma<<<gGemm, bGemm, 0, stream>>>(bfA,   Wt, bq + 1 * D_MODEL, 0, nullptr, Qbf);
    wt_transpose_bf16<<<gWt, 256, 0, stream>>>(Wk + 1 * DD, Wt);
    gemm_bf16_wmma<<<gGemm, bGemm, 0, stream>>>(zwsbf, Wt, bk + 1 * D_MODEL, 0, nullptr, Kbf);
    wt_transpose_bf16<<<gWt, 256, 0, stream>>>(Wv + 1 * DD, Wt);
    gemm_bf16_wmma<<<gGemm, bGemm, 0, stream>>>(zwsbf, Wt, bv + 1 * D_MODEL, 1, nullptr, Vtbf);
    flash_attn_wmma<<<gAttn, bAttn, 0, stream>>>(Qbf, Kbf, Vtbf, amask, Obf);
    wt_transpose_bf16<<<gWt, 256, 0, stream>>>(Wo + 1 * DD, Wt);
    gemm_bf16_wmma<<<gGemm, bGemm, 0, stream>>>(Obf, Wt, bo + 1 * D_MODEL, 2, zvw, nullptr);

    // ================= dual-LN residual fusion ==============================
    ln_fuse_kernel<<<NROWS, 256, 0, stream>>>(zvw, v_feat1, ln1g, ln1b, ln2g, ln2b, fusedbf);

    // ================= MHA block 2: self-attention on fused =================
    wt_transpose_bf16<<<gWt, 256, 0, stream>>>(Wq + 2 * DD, Wt);
    gemm_bf16_wmma<<<gGemm, bGemm, 0, stream>>>(fusedbf, Wt, bq + 2 * D_MODEL, 0, nullptr, Qbf);
    wt_transpose_bf16<<<gWt, 256, 0, stream>>>(Wk + 2 * DD, Wt);
    gemm_bf16_wmma<<<gGemm, bGemm, 0, stream>>>(fusedbf, Wt, bk + 2 * D_MODEL, 0, nullptr, Kbf);
    wt_transpose_bf16<<<gWt, 256, 0, stream>>>(Wv + 2 * DD, Wt);
    gemm_bf16_wmma<<<gGemm, bGemm, 0, stream>>>(fusedbf, Wt, bv + 2 * D_MODEL, 1, nullptr, Vtbf);
    flash_attn_wmma<<<gAttn, bAttn, 0, stream>>>(Qbf, Kbf, Vtbf, nullptr, Obf);
    wt_transpose_bf16<<<gWt, 256, 0, stream>>>(Wo + 2 * DD, Wt);
    gemm_bf16_wmma<<<gGemm, bGemm, 0, stream>>>(Obf, Wt, bo + 2 * D_MODEL, 2, out, nullptr);
}
